// RecurrentAttention_47356309406050
// MI455X (gfx1250) — matrix operand order repaired and checked by reading, not compiled, not run on hardware
//
#include <hip/hip_runtime.h>

typedef __bf16 bf16_t;
typedef __attribute__((ext_vector_type(16))) __bf16 v16bf;
typedef __attribute__((ext_vector_type(8)))  float  v8f;

#define BB   64
#define TT   64
#define SS   256
#define DIN  1024
#define HH   1024

// ---------------------------------------------------------------------------
// CDNA5 async global->LDS copy (ASYNCcnt-tracked, no VGPR data)
// ---------------------------------------------------------------------------
__device__ __forceinline__ void async_copy16(unsigned lds_byte_off, const void* gaddr) {
  asm volatile("global_load_async_to_lds_b128 %0, %1, off"
               :: "v"(lds_byte_off), "v"(gaddr) : "memory");
}
__device__ __forceinline__ void wait_async0() {
  asm volatile("s_wait_asynccnt 0x0" ::: "memory");
}

// ---------------------------------------------------------------------------
// fragment helpers (wave32; A and B share contiguous k = 16*(lane/16) + s, not the ISA table order)
// ---------------------------------------------------------------------------

// A fragment (16x32 bf16 tile) read from an LDS-staged [64][32] chunk.
// lane l holds row mt+(l&15); K = (l>=16 ? 16 : 0) + {0..15}
__device__ __forceinline__ v16bf lds_a_frag(const bf16_t* __restrict__ As, int mt, int lane) {
  int row = mt + (lane & 15);
  int k0  = (lane >> 4) << 4;
  const bf16_t* p = As + row * 32 + k0;
  v16bf f;
#pragma unroll
  for (int i = 0; i < 8; ++i) f[i] = p[i];          // 16B -> ds_load_b128
#pragma unroll
  for (int i = 0; i < 8; ++i) f[8 + i] = p[8 + i];  // 16B -> ds_load_b128
  return f;
}

// B[k][n] = W[n][k]  (computing X @ W^T, W row-major [N,K])
// lane l holds column nb+(l&15); K = kb + (l>=16 ? 16 : 0) + {0..15} (contiguous 32B)
__device__ __forceinline__ v16bf load_b_frag(const bf16_t* __restrict__ W, int ldw,
                                             int nb, int kb, int lane) {
  int col = nb + (lane & 15);
  int k0  = kb + ((lane >> 4) << 4);
  const bf16_t* p = W + (size_t)col * ldw + k0;
  v16bf f;
#pragma unroll
  for (int i = 0; i < 16; ++i) f[i] = p[i];
  return f;
}

__device__ __forceinline__ void store_c(float* __restrict__ C, int ldc, int mt, int nb,
                                        int lane, v8f acc, const float* __restrict__ bias) {
  int col = nb + (lane & 15);
  int r0  = mt + ((lane >> 4) << 3);
  float b = bias ? bias[col] : 0.0f;
#pragma unroll
  for (int i = 0; i < 8; ++i) C[(size_t)(r0 + i) * ldc + col] = acc[i] + b;
}

// ---------------------------------------------------------------------------
// C[64,N] = A[64,K](bf16) @ W[N,K]^T(bf16) + bias
// block = 256 threads = 8 waves; wave -> 64x16 tile; block -> 64x128 tile.
// A K-chunk [64x32] double-buffered in LDS, filled by async global->LDS loads
// (256 threads x 16B = 4KB per chunk); B fragment register-pipelined.
// REQUIRES: N % 128 == 0, K % 32 == 0 (all call sites satisfy this).
// ---------------------------------------------------------------------------
#define CHUNK_BYTES (64 * 32 * 2)   // one [64x32] bf16 buffer

__global__ void __launch_bounds__(256)
wmma_gemm_kernel(const bf16_t* __restrict__ A, const bf16_t* __restrict__ W,
                 const float* __restrict__ bias, float* __restrict__ C, int N, int K) {
  __shared__ alignas(16) bf16_t As[2 * 64 * 32];

  int tid  = threadIdx.x;
  int wave = tid >> 5;
  int lane = tid & 31;
  int nb   = (blockIdx.x * 8 + wave) * 16;   // wave-uniform, always < N

  // cooperative A-chunk fill: thread -> one b128 (row tid>>2, 8-elem segment)
  int arow = tid >> 2;                       // 0..63
  int aseg = (tid & 3) << 3;                 // 0,8,16,24
  const bf16_t* ag = A + (size_t)arow * K + aseg;
  // LDS aperture base is in addr[63:32]; low 32 bits == LDS byte offset
  unsigned lds_wr = (unsigned)(uintptr_t)(&As[arow * 32 + aseg]);

  // prologue: async-stage K-chunk 0 into buffer 0; preload first B fragment
  async_copy16(lds_wr, (const void*)ag);
  v16bf bcur = load_b_frag(W, K, nb, 0, lane);
  wait_async0();
  __syncthreads();

  v8f acc0 = {}, acc1 = {}, acc2 = {}, acc3 = {};
  int nsteps = K >> 5;
  unsigned buf = 0;                          // 0 / CHUNK_BYTES
  for (int s = 0; s < nsteps - 1; ++s) {
    int kb = s << 5;

    // fill the other buffer for K-step s+1 (overlaps with WMMAs below)
    async_copy16(lds_wr + (buf ^ CHUNK_BYTES), (const void*)(ag + kb + 32));
    v16bf bnext = load_b_frag(W, K, nb, kb + 32, lane);
    if (kb + 96 <= K) {
      // gfx1250 global_prefetch_b8 two K-chunks ahead on W
      __builtin_prefetch((const void*)(W + (size_t)(nb + (lane & 15)) * K + kb + 64), 0, 1);
    }

    // A fragments from the current LDS buffer (short-latency ds_load_b128)
    const bf16_t* asrc = As + (buf >> 1);    // byte offset -> element offset
    v16bf a0 = lds_a_frag(asrc,  0, lane);
    v16bf a1 = lds_a_frag(asrc, 16, lane);
    v16bf a2 = lds_a_frag(asrc, 32, lane);
    v16bf a3 = lds_a_frag(asrc, 48, lane);

    acc0 = __builtin_amdgcn_wmma_f32_16x16x32_bf16(false, a0, false, bcur, (short)0, acc0, false, false);
    acc1 = __builtin_amdgcn_wmma_f32_16x16x32_bf16(false, a1, false, bcur, (short)0, acc1, false, false);
    acc2 = __builtin_amdgcn_wmma_f32_16x16x32_bf16(false, a2, false, bcur, (short)0, acc2, false, false);
    acc3 = __builtin_amdgcn_wmma_f32_16x16x32_bf16(false, a3, false, bcur, (short)0, acc3, false, false);

    bcur = bnext;
    wait_async0();                           // our async fill has landed in LDS
    __syncthreads();                         // publish across all 8 waves
    buf ^= CHUNK_BYTES;
  }

  // peeled final K-step (no staging)
  {
    const bf16_t* asrc = As + (buf >> 1);
    v16bf a0 = lds_a_frag(asrc,  0, lane);
    v16bf a1 = lds_a_frag(asrc, 16, lane);
    v16bf a2 = lds_a_frag(asrc, 32, lane);
    v16bf a3 = lds_a_frag(asrc, 48, lane);
    acc0 = __builtin_amdgcn_wmma_f32_16x16x32_bf16(false, a0, false, bcur, (short)0, acc0, false, false);
    acc1 = __builtin_amdgcn_wmma_f32_16x16x32_bf16(false, a1, false, bcur, (short)0, acc1, false, false);
    acc2 = __builtin_amdgcn_wmma_f32_16x16x32_bf16(false, a2, false, bcur, (short)0, acc2, false, false);
    acc3 = __builtin_amdgcn_wmma_f32_16x16x32_bf16(false, a3, false, bcur, (short)0, acc3, false, false);
  }

  store_c(C, N,  0, nb, lane, acc0, bias);
  store_c(C, N, 16, nb, lane, acc1, bias);
  store_c(C, N, 32, nb, lane, acc2, bias);
  store_c(C, N, 48, nb, lane, acc3, bias);
}

// ---------------------------------------------------------------------------
// weight prep: dst row n = [Wa[n,:Ka] | Wb[n,:Kb]] as bf16
// ---------------------------------------------------------------------------
__global__ void cvt_concat_kernel(const float* __restrict__ Wa, int Ka,
                                  const float* __restrict__ Wb, int Kb,
                                  bf16_t* __restrict__ dst, long total) {
  long idx = (long)blockIdx.x * blockDim.x + threadIdx.x;
  if (idx >= total) return;
  int K = Ka + Kb;
  long n = idx / K;
  int  k = (int)(idx - n * K);
  float v = (k < Ka) ? Wa[n * Ka + k] : Wb[n * Kb + (k - Ka)];
  dst[idx] = (bf16_t)v;
}

__global__ void vec_add_kernel(const float* a, const float* b, float* d, int n) {
  int i = blockIdx.x * blockDim.x + threadIdx.x;
  if (i < n) d[i] = a[i] + b[i];
}

__global__ void zero_kernel(unsigned int* p, long n_words) {
  long i = (long)blockIdx.x * blockDim.x + threadIdx.x;
  if (i < n_words) p[i] = 0u;
}

// xin0[:, 0:1024] = bf16(inputs[:, t, :])
__global__ void pack_x_kernel(const float* __restrict__ inputs, bf16_t* __restrict__ xin0, int t) {
  int idx = blockIdx.x * blockDim.x + threadIdx.x;     // 64*1024
  int b = idx >> 10, j = idx & 1023;
  xin0[b * 3072 + j] = (bf16_t)inputs[((size_t)b * TT + t) * DIN + j];
}

__device__ __forceinline__ float sigmoid_f(float x) { return 1.0f / (1.0f + __expf(-x)); }

// PyTorch gate order i,f,g,o; writes h as bf16 to up to 3 destinations
__global__ void lstm_gates_kernel(const float* __restrict__ G, float* __restrict__ c,
                                  bf16_t* d1, int s1, bf16_t* d2, int s2, bf16_t* d3, int s3) {
  int idx = blockIdx.x * blockDim.x + threadIdx.x;     // 64*1024
  int b = idx >> 10, j = idx & 1023;
  const float* g = G + (size_t)b * 4 * HH;
  float ig = sigmoid_f(g[j]);
  float fg = sigmoid_f(g[HH + j]);
  float gg = tanhf(g[2 * HH + j]);
  float og = sigmoid_f(g[3 * HH + j]);
  float cv = fg * c[idx] + ig * gg;
  c[idx] = cv;
  float h = og * tanhf(cv);
  bf16_t hb = (bf16_t)h;
  if (d1) d1[b * s1 + j] = hb;
  if (d2) d2[b * s2 + j] = hb;
  if (d3) d3[b * s3 + j] = hb;
}

// per-batch attention: scores = q . ctx[b,s,:], softmax over s, mixed -> xo[:, 0:1024]
__global__ void __launch_bounds__(256)
attn_kernel(const float* __restrict__ q, const float* __restrict__ ctx, bf16_t* __restrict__ xo) {
  int b = blockIdx.x;
  int tid = threadIdx.x, wave = tid >> 5, lane = tid & 31;
  __shared__ float sc[SS];
  __shared__ float red[SS];
  const float* qb = q + (size_t)b * HH;
  const float* cb = ctx + (size_t)b * SS * HH;

  for (int i = 0; i < 32; ++i) {                 // 8 waves x 32 scores
    int s = wave * 32 + i;
    const float* row = cb + (size_t)s * HH;
    float d = 0.0f;
    for (int h = lane; h < HH; h += 32) d += qb[h] * row[h];
#pragma unroll
    for (int off = 16; off > 0; off >>= 1) d += __shfl_xor(d, off, 32);
    if (lane == 0) sc[s] = d;
  }
  __syncthreads();

  float v = sc[tid];
  red[tid] = v; __syncthreads();
  for (int off = 128; off > 0; off >>= 1) {
    if (tid < off) red[tid] = fmaxf(red[tid], red[tid + off]);
    __syncthreads();
  }
  float m = red[0]; __syncthreads();
  float e = __expf(v - m);
  red[tid] = e; __syncthreads();
  for (int off = 128; off > 0; off >>= 1) {
    if (tid < off) red[tid] += red[tid + off];
    __syncthreads();
  }
  float inv = 1.0f / red[0];
  __syncthreads();
  sc[tid] = e * inv;
  __syncthreads();

  for (int h = tid; h < HH; h += 256) {
    float acc = 0.0f;
    for (int s = 0; s < SS; ++s) acc += sc[s] * cb[(size_t)s * HH + h];
    xo[b * 2048 + h] = (bf16_t)acc;
  }
}

// out[b,t,:] = tanh(tmp); also feed a_{t} (bf16) into next step's xin0[:,1024:2048]
__global__ void tanh_store_kernel(const float* __restrict__ tmp, float* __restrict__ out,
                                  bf16_t* __restrict__ a_dst, int t) {
  int idx = blockIdx.x * blockDim.x + threadIdx.x;     // 64*1024
  int b = idx >> 10, j = idx & 1023;
  float v = tanhf(tmp[idx]);
  out[((size_t)b * TT + t) * HH + j] = v;
  a_dst[b * 3072 + j] = (bf16_t)v;
}

// ---------------------------------------------------------------------------
extern "C" void kernel_launch(void* const* d_in, const int* in_sizes, int n_in,
                              void* d_out, int out_size, void* d_ws, size_t ws_size,
                              hipStream_t stream) {
  const float* inputs  = (const float*)d_in[0];
  const float* context = (const float*)d_in[1];
  const float* Wi0 = (const float*)d_in[2];
  const float* Wh0 = (const float*)d_in[3];
  const float* bi0 = (const float*)d_in[4];
  const float* bh0 = (const float*)d_in[5];
  const float* Wi1 = (const float*)d_in[6];
  const float* Wh1 = (const float*)d_in[7];
  const float* bi1 = (const float*)d_in[8];
  const float* bh1 = (const float*)d_in[9];
  const float* Wq  = (const float*)d_in[10];
  const float* Wo  = (const float*)d_in[11];
  const float* bo  = (const float*)d_in[12];
  float* out = (float*)d_out;

  char* p = (char*)d_ws;
  auto alloc = [&](size_t bytes) { char* r = p; p += (bytes + 255) & ~(size_t)255; return r; };

  bf16_t* w0  = (bf16_t*)alloc((size_t)4096 * 3072 * 2);   // [Wi0|Wh0]
  bf16_t* w1  = (bf16_t*)alloc((size_t)4096 * 2048 * 2);   // [Wi1|Wh1]
  bf16_t* wq  = (bf16_t*)alloc((size_t)1024 * 1024 * 2);
  bf16_t* wo  = (bf16_t*)alloc((size_t)1024 * 2048 * 2);
  float*  b0  = (float*)alloc(4096 * 4);
  float*  b1  = (float*)alloc(4096 * 4);
  float*  G   = (float*)alloc((size_t)64 * 4096 * 4);      // gate pre-activations
  float*  qbf = (float*)alloc((size_t)64 * 1024 * 4);
  float*  tmpo= (float*)alloc((size_t)64 * 1024 * 4);
  // zero-init region (contiguous): c0,c1,xin0,xin1,xq,xo
  char*   zbase = p;
  float*  c0   = (float*)alloc((size_t)64 * 1024 * 4);
  float*  c1   = (float*)alloc((size_t)64 * 1024 * 4);
  bf16_t* xin0 = (bf16_t*)alloc((size_t)64 * 3072 * 2);    // [x_t | a | h0]
  bf16_t* xin1 = (bf16_t*)alloc((size_t)64 * 2048 * 2);    // [h0 | h1]
  bf16_t* xq   = (bf16_t*)alloc((size_t)64 * 1024 * 2);    // h1
  bf16_t* xo   = (bf16_t*)alloc((size_t)64 * 2048 * 2);    // [mixed | h1]
  long zero_words = (long)(p - zbase) / 4;

  // ---- one-time (per launch) prep ----
  {
    long n0 = (long)4096 * 3072;
    cvt_concat_kernel<<<(unsigned)((n0 + 255) / 256), 256, 0, stream>>>(Wi0, 2048, Wh0, 1024, w0, n0);
    long n1 = (long)4096 * 2048;
    cvt_concat_kernel<<<(unsigned)((n1 + 255) / 256), 256, 0, stream>>>(Wi1, 1024, Wh1, 1024, w1, n1);
    long nq = (long)1024 * 1024;
    cvt_concat_kernel<<<(unsigned)((nq + 255) / 256), 256, 0, stream>>>(Wq, 1024, (const float*)nullptr, 0, wq, nq);
    long no = (long)1024 * 2048;
    cvt_concat_kernel<<<(unsigned)((no + 255) / 256), 256, 0, stream>>>(Wo, 2048, (const float*)nullptr, 0, wo, no);
    vec_add_kernel<<<16, 256, 0, stream>>>(bi0, bh0, b0, 4096);
    vec_add_kernel<<<16, 256, 0, stream>>>(bi1, bh1, b1, 4096);
    zero_kernel<<<(unsigned)((zero_words + 255) / 256), 256, 0, stream>>>((unsigned int*)zbase, zero_words);
  }

  const int EW = (64 * 1024) / 256;  // elementwise grid (256 blocks)

  for (int t = 0; t < TT; ++t) {
    // xin0 = [x_t, a(prev), h0(prev)]  (a, h0 segments written by prev step)
    pack_x_kernel<<<EW, 256, 0, stream>>>(inputs, xin0, t);
    // cell 0: G = xin0 @ [Wi0|Wh0]^T + (bi0+bh0)
    wmma_gemm_kernel<<<4096 / 128, 256, 0, stream>>>(xin0, w0, b0, G, 4096, 3072);
    // h0 -> xin1[:,0:1024] (this step's cell1), xin0[:,2048:3072] (next step)
    lstm_gates_kernel<<<EW, 256, 0, stream>>>(G, c0, xin1, 2048, xin0 + 2048, 3072,
                                              (bf16_t*)nullptr, 0);
    // cell 1: G = [h0,h1] @ [Wi1|Wh1]^T + (bi1+bh1)
    wmma_gemm_kernel<<<4096 / 128, 256, 0, stream>>>(xin1, w1, b1, G, 4096, 2048);
    // h1 -> xq (this step's Wq), xo[:,1024:2048] (this step's Wo), xin1[:,1024:] (next step)
    lstm_gates_kernel<<<EW, 256, 0, stream>>>(G, c1, xq, 1024, xo + 1024, 2048, xin1 + 1024, 2048);
    // q = h1 @ Wq^T
    wmma_gemm_kernel<<<1024 / 128, 256, 0, stream>>>(xq, wq, (const float*)nullptr, qbf, 1024, 1024);
    // attention -> mixed into xo[:,0:1024]
    attn_kernel<<<BB, 256, 0, stream>>>(qbf, context, xo);
    // tmpo = [mixed,h1] @ Wo^T + bo
    wmma_gemm_kernel<<<1024 / 128, 256, 0, stream>>>(xo, wo, bo, tmpo, 1024, 2048);
    // a = tanh(tmpo) -> d_out[:,t,:] and xin0[:,1024:2048] for next step
    tanh_store_kernel<<<EW, 256, 0, stream>>>(tmpo, out, xin0 + 1024, t);
  }
}
